// RNN_31636729102350
// MI455X (gfx1250) — matrix operand order repaired and checked by reading, not compiled, not run on hardware
//
#include <hip/hip_runtime.h>
#include <hip/hip_bf16.h>
#include <math.h>

// Problem dimensions (fixed by the reference).
#define Vv 50000
#define Ee 512
#define Hh 512
#define Bb 64
#define Ss 512
#define G3 (3 * Hh)   // 1536
#define NWG 32        // persistent workgroups for the recurrent scan

typedef __attribute__((ext_vector_type(16))) __bf16 v16bf;
typedef __attribute__((ext_vector_type(8)))  float  v8f;

union Frag { v16bf v; uint4 u[2]; };

__device__ __forceinline__ v8f wmma_bf16(v16bf a, v16bf b, v8f c) {
  // D = A(16x32 bf16) * B(32x16 bf16) + C(16x16 f32)
  return __builtin_amdgcn_wmma_f32_16x16x32_bf16(false, a, false, b,
                                                 (short)0, c, false, false);
}

__device__ __forceinline__ __bf16 f2bf(float f) {
  unsigned u = __float_as_uint(f);
  unsigned r = (u + 0x7FFFu + ((u >> 16) & 1u)) >> 16;  // RNE
  unsigned short s = (unsigned short)r;
  return *reinterpret_cast<__bf16*>(&s);
}
__device__ __forceinline__ float bf2f(__bf16 b) {
  unsigned short s = *reinterpret_cast<unsigned short*>(&b);
  unsigned u = ((unsigned)s) << 16;
  return __uint_as_float(u);
}
__device__ __forceinline__ float sigmoidf_(float x) {
  return 1.f / (1.f + __expf(-x));
}

// ---------------------------------------------------------------------------
// fp32 -> bf16 conversion (emb, W_ih, W_hh). Pure bandwidth, trivial.
// ---------------------------------------------------------------------------
__global__ void cvt_bf16_kernel(const float* __restrict__ in,
                                __bf16* __restrict__ out, int n) {
  int i = blockIdx.x * blockDim.x + threadIdx.x;
  if (i < n) out[i] = f2bf(in[i]);
}

// ---------------------------------------------------------------------------
// Zero the grid-barrier counter and h0 every launch (determinism).
// ---------------------------------------------------------------------------
__global__ void init_kernel(unsigned* __restrict__ hbuf0_u32,
                            unsigned* __restrict__ cnt) {
  int i = blockIdx.x * blockDim.x + threadIdx.x;
  if (i < (Bb * Hh * 2) / 4) hbuf0_u32[i] = 0u;   // h[0] = 0 (bf16 zeros)
  if (i == 0) *cnt = 0u;
}

// ---------------------------------------------------------------------------
// Kernel A: xp[s*B+b, :] = emb[x[b,s]] @ W_ih^T + b_ih   (32768 x 1536 x 512)
// bf16 WMMA, fp32 accumulate. Block = 256 thr (8 waves), tile 64x128,
// each wave: 2x2 grid of 16x16 accumulators, K-loop of 16 x (K=32).
// Fragments loaded straight from global (emb/W_ih are L2-resident bf16).
// ---------------------------------------------------------------------------
__global__ __launch_bounds__(256)
void xproj_kernel(const int* __restrict__ x, const __bf16* __restrict__ embb,
                  const __bf16* __restrict__ wihb, const float* __restrict__ bih,
                  float* __restrict__ xp) {
  const int l  = threadIdx.x & 31;
  const int w  = threadIdx.x >> 5;                 // 0..7
  const int m0 = blockIdx.x * 64 + (w & 1) * 32;   // row base for this wave
  const int n0 = blockIdx.y * 128 + (w >> 1) * 32; // col base for this wave
  const int akh = (l >> 4) << 4;                   // A: K half-offset 0 / 16
  const int bkh = (l >> 4) << 4;                   // B: K half-offset 0 / 16

  // Per-lane A row pointers (gathered embedding rows).
  const __bf16* arow[2];
#pragma unroll
  for (int mt = 0; mt < 2; ++mt) {
    int row = m0 + mt * 16 + (l & 15);             // row = s*64 + b
    int s = row >> 6, b = row & 63;
    int tok = x[b * Ss + s];
    arow[mt] = embb + (size_t)tok * Ee;
  }
  // Per-lane B column pointers: B[k,n] = W_ih[n,k] (row-major [N,K]).
  const __bf16* brow[2];
#pragma unroll
  for (int nt = 0; nt < 2; ++nt) {
    int n = n0 + nt * 16 + (l & 15);
    brow[nt] = wihb + (size_t)n * Ee;
  }

  v8f acc[2][2] = {{{0}}};
  for (int k = 0; k < Ee; k += 32) {
    Frag a[2], bfr[2];
#pragma unroll
    for (int mt = 0; mt < 2; ++mt) {
      a[mt].u[0] = *(const uint4*)(arow[mt] + k + akh);
      a[mt].u[1] = *(const uint4*)(arow[mt] + k + akh + 8);
    }
#pragma unroll
    for (int nt = 0; nt < 2; ++nt) {
      bfr[nt].u[0] = *(const uint4*)(brow[nt] + k + bkh);
      bfr[nt].u[1] = *(const uint4*)(brow[nt] + k + bkh + 8);
    }
#pragma unroll
    for (int mt = 0; mt < 2; ++mt)
#pragma unroll
      for (int nt = 0; nt < 2; ++nt)
        acc[mt][nt] = wmma_bf16(a[mt].v, bfr[nt].v, acc[mt][nt]);
  }

  // Epilogue: +bias, store fp32. C layout: N = lane%16, M = vgpr + 8*(lane>=16).
#pragma unroll
  for (int mt = 0; mt < 2; ++mt) {
    int rowb = m0 + mt * 16 + ((l >> 4) << 3);
#pragma unroll
    for (int nt = 0; nt < 2; ++nt) {
      int n = n0 + nt * 16 + (l & 15);
      float bias = bih[n];
#pragma unroll
      for (int r = 0; r < 8; ++r)
        xp[(size_t)(rowb + r) * G3 + n] = acc[mt][nt][r] + bias;
    }
  }
}

// ---------------------------------------------------------------------------
// Kernel B: persistent GRU scan. 32 WGs x 128 thr; WG g owns hidden columns
// [16g, 16g+16); wave w owns batch rows [16w, 16w+16).
//
// Latency optimization for the serial chain: this WG's W_hh slice
// (3 gates x 16 cols x 512 K = 48 KB bf16) is staged ONCE into LDS in the
// exact per-lane WMMA B-fragment layout (32 B per (gate, kchunk, lane)), so
// every step's K-loop reads B via ds_load_b128 (LDS latency) instead of
// global/L2. Per-step gate inputs xp[t] and h_prev are issued at the top of
// the step so they overlap the WMMA loop. h is double-buffered in global
// (L2-resident, 64 KB) with one device-wide barrier per step.
// ---------------------------------------------------------------------------
__global__ __launch_bounds__(128)
void gru_scan_kernel(const float* __restrict__ xp, const __bf16* __restrict__ whhb,
                     const float* __restrict__ bhh, __bf16* __restrict__ hbuf,
                     float* __restrict__ hT, unsigned* __restrict__ cnt) {
  __shared__ uint4 ldsb[3 * 16 * 32 * 2];        // 49152 B: [gate][kc][lane][2]

  const int l  = threadIdx.x & 31;
  const int w  = threadIdx.x >> 5;     // 0..3
  const int j0 = blockIdx.x * 16;      // hidden-column base for this WG
  const int jl = j0 + (l & 15);        // this lane's hidden column (N = lane%16)
  const int akh = (l >> 4) << 4;
  const int bkh = (l >> 4) << 4;

  // Stage W_hh B-fragments into LDS (waves split the 16 K-chunks).
  {
    const __bf16* pr = whhb + (size_t)jl * Hh;
    const __bf16* pz = whhb + (size_t)(Hh + jl) * Hh;
    const __bf16* pn = whhb + (size_t)(2 * Hh + jl) * Hh;
    for (int kc = w; kc < 16; kc += 4) {
      int k = kc * 32 + bkh;
      int s0 = ((0 * 16 + kc) * 32 + l) * 2;
      int s1 = ((1 * 16 + kc) * 32 + l) * 2;
      int s2 = ((2 * 16 + kc) * 32 + l) * 2;
      ldsb[s0]     = *(const uint4*)(pr + k);
      ldsb[s0 + 1] = *(const uint4*)(pr + k + 8);
      ldsb[s1]     = *(const uint4*)(pz + k);
      ldsb[s1 + 1] = *(const uint4*)(pz + k + 8);
      ldsb[s2]     = *(const uint4*)(pn + k);
      ldsb[s2 + 1] = *(const uint4*)(pn + k + 8);
    }
  }
  __syncthreads();

  const float bhr = bhh[jl], bhz = bhh[Hh + jl], bhn = bhh[2 * Hh + jl];
  const int arow = 16 * w + (l & 15);            // A-load batch row
  const int rowb = 16 * w + ((l >> 4) << 3);     // C-layout batch row base

  for (int t = 0; t < Ss; ++t) {
    const __bf16* hin  = hbuf + (size_t)(t & 1) * (Bb * Hh);
    __bf16*       hout = hbuf + (size_t)((t + 1) & 1) * (Bb * Hh);

    // Prefetch gate inputs + previous h for this lane's 8 output elements
    // (overlaps with the WMMA K-loop below).
    float xr[8], xz[8], xn[8], hp[8];
    const float* xprow = xp + (size_t)t * Bb * G3;
#pragma unroll
    for (int r = 0; r < 8; ++r) {
      const float* xpe = xprow + (size_t)(rowb + r) * G3;
      xr[r] = xpe[jl];
      xz[r] = xpe[Hh + jl];
      xn[r] = xpe[2 * Hh + jl];
      hp[r] = bf2f(hin[(size_t)(rowb + r) * Hh + jl]);
    }

    // hp_{r,z,n}[16x16] = h[16 rows x 512] @ W_hh_gate^T[512 x 16 cols]
    v8f ar = {0}, az = {0}, an = {0};
    const __bf16* ap = hin + (size_t)arow * Hh + akh;
#pragma unroll 4
    for (int kc = 0; kc < 16; ++kc) {
      Frag a, fr, fz, fn;
      a.u[0] = *(const uint4*)(ap + kc * 32);
      a.u[1] = *(const uint4*)(ap + kc * 32 + 8);
      int s0 = ((0 * 16 + kc) * 32 + l) * 2;
      int s1 = ((1 * 16 + kc) * 32 + l) * 2;
      int s2 = ((2 * 16 + kc) * 32 + l) * 2;
      fr.u[0] = ldsb[s0]; fr.u[1] = ldsb[s0 + 1];
      fz.u[0] = ldsb[s1]; fz.u[1] = ldsb[s1 + 1];
      fn.u[0] = ldsb[s2]; fn.u[1] = ldsb[s2 + 1];
      ar = wmma_bf16(a.v, fr.v, ar);
      az = wmma_bf16(a.v, fz.v, az);
      an = wmma_bf16(a.v, fn.v, an);
    }

    // Fused gates: r = sig(xr+hr), z = sig(xz+hz), n = tanh(xn + r*hn),
    // h' = (1-z)*n + z*h. Element (row = rowb+r, col = jl).
#pragma unroll
    for (int r = 0; r < 8; ++r) {
      float rg = sigmoidf_(xr[r] + ar[r] + bhr);
      float zg = sigmoidf_(xz[r] + az[r] + bhz);
      float ng = tanhf(xn[r] + rg * (an[r] + bhn));
      float hn2 = (1.f - zg) * ng + zg * hp[r];
      hout[(size_t)(rowb + r) * Hh + jl] = f2bf(hn2);
      if (t == Ss - 1) hT[(size_t)(rowb + r) * Hh + jl] = hn2;
    }

    // Device-wide step barrier (monotonic counter, re-zeroed by init_kernel).
    __syncthreads();
    if (threadIdx.x == 0) {
      __threadfence();                 // release h writes to device scope
      atomicAdd(cnt, 1u);
      unsigned target = (unsigned)(t + 1) * NWG;
      while (__hip_atomic_load(cnt, __ATOMIC_RELAXED,
                               __HIP_MEMORY_SCOPE_AGENT) < target)
        __builtin_amdgcn_s_sleep(2);
    }
    __syncthreads();
    __threadfence();                   // acquire: invalidate near caches
  }
}

// ---------------------------------------------------------------------------
// Kernel C: out[b] = sigmoid(hT[b,:] . fc_w + fc_b). One wave per batch row.
// ---------------------------------------------------------------------------
__global__ __launch_bounds__(32)
void fc_kernel(const float* __restrict__ hT, const float* __restrict__ fcw,
               const float* __restrict__ fcb, float* __restrict__ out) {
  int b = blockIdx.x;
  int l = threadIdx.x;
  float s = 0.f;
  for (int i = l; i < Hh; i += 32) s += hT[(size_t)b * Hh + i] * fcw[i];
#pragma unroll
  for (int off = 16; off; off >>= 1) s += __shfl_xor(s, off, 32);
  if (l == 0) out[b] = sigmoidf_(s + fcb[0]);
}

// ---------------------------------------------------------------------------
// Host-side launch. Workspace layout (bytes, 256-aligned chunks):
//   xp   fp32 [S*B, 3H] : 201326592
//   embb bf16 [V, E]    :  51200000
//   wihb bf16 [3H, E]   :   1572864
//   whhb bf16 [3H, H]   :   1572864
//   hbuf bf16 [2, B*H]  :    131072
//   hT   fp32 [B, H]    :    131072
//   cnt  u32            :       256     total ~244 MiB
// ---------------------------------------------------------------------------
extern "C" void kernel_launch(void* const* d_in, const int* in_sizes, int n_in,
                              void* d_out, int out_size, void* d_ws, size_t ws_size,
                              hipStream_t stream) {
  const int*   x    = (const int*)d_in[0];
  const float* emb  = (const float*)d_in[1];
  const float* wih  = (const float*)d_in[2];
  const float* bih  = (const float*)d_in[3];
  const float* whh  = (const float*)d_in[4];
  const float* bhh  = (const float*)d_in[5];
  const float* fcw  = (const float*)d_in[6];
  const float* fcb  = (const float*)d_in[7];
  float* out = (float*)d_out;

  char* ws = (char*)d_ws;
  size_t off = 0;
  float*   xp   = (float*)(ws + off);   off += (size_t)Ss * Bb * G3 * 4;  // 201326592
  __bf16*  embb = (__bf16*)(ws + off);  off += (size_t)Vv * Ee * 2;       //  51200000
  __bf16*  wihb = (__bf16*)(ws + off);  off += (size_t)G3 * Ee * 2;       //   1572864
  __bf16*  whhb = (__bf16*)(ws + off);  off += (size_t)G3 * Hh * 2;       //   1572864
  __bf16*  hbuf = (__bf16*)(ws + off);  off += (size_t)2 * Bb * Hh * 2;   //    131072
  float*   hT   = (float*)(ws + off);   off += (size_t)Bb * Hh * 4;       //    131072
  unsigned* cnt = (unsigned*)(ws + off);

  // fp32 -> bf16 operand conversion.
  int ne = Vv * Ee, nw = G3 * Ee;
  cvt_bf16_kernel<<<(ne + 255) / 256, 256, 0, stream>>>(emb, embb, ne);
  cvt_bf16_kernel<<<(nw + 255) / 256, 256, 0, stream>>>(wih, wihb, nw);
  cvt_bf16_kernel<<<(nw + 255) / 256, 256, 0, stream>>>(whh, whhb, nw);

  // Re-zero barrier counter + h0 (deterministic, graph-safe).
  init_kernel<<<64, 256, 0, stream>>>((unsigned*)hbuf, cnt);

  // Big input-projection GEMM: grid (M/64, N/128).
  xproj_kernel<<<dim3((Ss * Bb) / 64, G3 / 128), 256, 0, stream>>>(
      x, embb, wihb, bih, xp);

  // Persistent recurrent scan.
  gru_scan_kernel<<<NWG, 128, 0, stream>>>(xp, whhb, bhh, hbuf, hT, cnt);

  // Final FC + sigmoid.
  fc_kernel<<<Bb, 32, 0, stream>>>(hT, fcw, fcb, out);
}
